// QSASRecD_16415365005505
// MI455X (gfx1250) — matrix operand order repaired and checked by reading, not compiled, not run on hardware
//
#include <hip/hip_runtime.h>
#include <hip/hip_bf16.h>

// ---------------------------------------------------------------------------
// CDNA5 (gfx1250, wave32) WMMA implementation of the QSASRec attention block.
// All GEMMs and the attention score/context products use
// v_wmma_f32_16x16x32_f16. f32 inputs are converted to f16; accumulation f32.
//
// VGPR fragment layouts per CDNA5 ISA:
//   A (16x32 f16):  lane m=l&15; klo=(l>=16)?8:0; elems[0..7]=K klo..klo+7,
//                   elems[8..15]=K klo+16..klo+23  (two contiguous 16B loads)
//   B (32x16 f16):  lane n=l&15; klo=(l>=16)?8:0; same k order as A
//                   (two contiguous 16B loads)
//   C/D (16x16 f32): lane owns column n=l&15, rows (l>=16?8:0)+j, j=vgpr 0..7
// ---------------------------------------------------------------------------

typedef __attribute__((ext_vector_type(16))) _Float16 v16h;
typedef __attribute__((ext_vector_type(8)))  _Float16 v8h;
typedef __attribute__((ext_vector_type(8)))  float    v8f;

#if __has_builtin(__builtin_amdgcn_global_load_async_to_lds_b128) && \
    __has_builtin(__builtin_amdgcn_s_wait_asynccnt)
#define HAVE_ASYNC_LDS 1
// The builtin wants addrspace-qualified v4i pointers:
//   param0: int __vector(4) addrspace(1)* (prints as "__device__" in HIP)
//   param1: LDS destination, addrspace(3)
typedef int v4i_ __attribute__((__vector_size__(16)));
typedef __attribute__((address_space(1))) v4i_ gas_v4i;
typedef __attribute__((address_space(3))) v4i_ las_v4i;
#else
#define HAVE_ASYNC_LDS 0
#endif

union AFrag { v16h v; v8h h[2]; _Float16 e[16]; };
union CFrag { v8f  v; float f[8]; };

__device__ __forceinline__ v8f wmma_f16(v16h a, v16h b, v8f c) {
    return __builtin_amdgcn_wmma_f32_16x16x32_f16(
        /*neg_a=*/false, a, /*neg_b=*/false, b,
        /*c_mod=*/(short)0, c, /*reuse_a=*/false, /*reuse_b=*/false);
}

// A fragment from a row-major f16 matrix (row stride in elements).
__device__ __forceinline__ v16h load_afrag(const _Float16* rowbase, int stride,
                                           int chunkBase, int lane) {
    const int m16 = lane & 15;
    const int klo = (lane >> 4) * 8;
    const _Float16* r = rowbase + (size_t)m16 * stride + chunkBase + klo;
    AFrag f;
    f.h[0] = *(const v8h*)(r);
    f.h[1] = *(const v8h*)(r + 16);
    return f.v;
}

// B fragment where B = M^T and M is row-major [n][k-dim] (e.g. K, KP, KA):
// lane n reads features featBase+hi*8..+7 and +16..+23 from row (rowBase + n).
__device__ __forceinline__ v16h load_bfrag_rows(const _Float16* base, int rowBase,
                                                int stride, int featBase, int lane) {
    const int m16 = lane & 15;
    const int hi  = lane >> 4;
    const _Float16* r = base + (size_t)(rowBase + m16) * stride + featBase + hi * 8;
    AFrag f;
    f.h[0] = *(const v8h*)(r);
    f.h[1] = *(const v8h*)(r + 16);
    return f.v;
}

// ------------------------------ weight convert ------------------------------
__global__ void cvt_f32_f16(const float* __restrict__ src,
                            _Float16* __restrict__ dst, int n) {
    int i = blockIdx.x * blockDim.x + threadIdx.x;
    if (i < n) dst[i] = (_Float16)src[i];
}

// ------------------------------- GEMM (proj) --------------------------------
// Y[M,N] = X[M,K] @ W[K,N] + bias.  Block: 64x64 tile, 256 threads = 8 waves,
// each wave one 16x32 strip (2 C tiles).  K stepped by 32 via LDS staging.
// MODE 0: store f16 head-split: out[((b*4+h)*1024+s)*outPitch + outOff + d]
// MODE 1: store f32 row-major out[r*N + n]
// XISHALF selects the X dtype at compile time (keeps the k-loop branch-free);
// the f16 path uses GLOBAL_LOAD_ASYNC_TO_LDS_B128 when the builtin exists.
template <int XISHALF, int MODE>
__global__ __launch_bounds__(256) void gemm_proj(
    const float* __restrict__ Xf, const _Float16* __restrict__ Xh,
    const _Float16* __restrict__ W, const float* __restrict__ bias,
    _Float16* __restrict__ outH, float* __restrict__ outF,
    int M, int N, int K, int headDim, int outPitch, int outOff)
{
    __shared__ __align__(16) _Float16 Xs[64][32];   // [row][k]
    __shared__ __align__(16) _Float16 Ws[64][32];   // [n][k] (transposed)

    const int tid  = threadIdx.x;
    const int lane = tid & 31;
    const int wave = tid >> 5;
    const int m16  = lane & 15;
    const int hi   = lane >> 4;
    const int wr   = wave >> 1;           // 0..3 row tile
    const int wc   = wave & 1;            // 0..1 col half
    const int rowBase = blockIdx.y * 64;
    const int colBase = blockIdx.x * 64;

    const int xr  = tid >> 2;             // 0..63  X staging row
    const int xks = (tid & 3) * 8;        // 0..24  X staging k-seg
    const int wk  = tid >> 3;             // 0..31  W staging k
    const int wns = (tid & 7) * 8;        // 0..56  W staging n-seg

    CFrag acc[2];
#pragma unroll
    for (int t = 0; t < 2; ++t)
#pragma unroll
        for (int j = 0; j < 8; ++j) acc[t].f[j] = 0.0f;

    for (int kb = 0; kb < K; kb += 32) {
        // ---- stage X tile (convert f32->f16 if needed) ----
        if (XISHALF) {
            const _Float16* p = Xh + (size_t)(rowBase + xr) * K + kb + xks;
#if HAVE_ASYNC_LDS
            // direct global->LDS DMA, tracked with ASYNCcnt
            __builtin_amdgcn_global_load_async_to_lds_b128(
                (gas_v4i*)p, (las_v4i*)&Xs[xr][xks], 0, 0);
#else
            *(v8h*)&Xs[xr][xks] = *(const v8h*)p;
#endif
        } else {
            const float* p = Xf + (size_t)(rowBase + xr) * K + kb + xks;
            float4 f0 = *(const float4*)(p);
            float4 f1 = *(const float4*)(p + 4);
            v8h hv;
            hv[0] = (_Float16)f0.x; hv[1] = (_Float16)f0.y;
            hv[2] = (_Float16)f0.z; hv[3] = (_Float16)f0.w;
            hv[4] = (_Float16)f1.x; hv[5] = (_Float16)f1.y;
            hv[6] = (_Float16)f1.z; hv[7] = (_Float16)f1.w;
            *(v8h*)&Xs[xr][xks] = hv;
            if (kb + 32 < K) __builtin_prefetch(p + 32, 0, 1);  // global_prefetch_b8
        }
        // ---- stage W tile transposed ----
        {
            const _Float16* p = W + (size_t)(kb + wk) * N + colBase + wns;
            v8h wv = *(const v8h*)p;
#pragma unroll
            for (int i = 0; i < 8; ++i) Ws[wns + i][wk] = wv[i];
        }
#if HAVE_ASYNC_LDS
        if (XISHALF) __builtin_amdgcn_s_wait_asynccnt(0);
#endif
        __syncthreads();

        AFrag a;
        {
            const _Float16* r = &Xs[wr * 16 + m16][hi * 8];
            a.h[0] = *(const v8h*)(r);
            a.h[1] = *(const v8h*)(r + 16);
        }
#pragma unroll
        for (int ct = 0; ct < 2; ++ct) {
            AFrag b;
            const _Float16* r = &Ws[wc * 32 + ct * 16 + m16][hi * 8];
            b.h[0] = *(const v8h*)(r);
            b.h[1] = *(const v8h*)(r + 16);
            acc[ct].v = wmma_f16(a.v, b.v, acc[ct].v);
        }
        __syncthreads();
    }

    // ---- epilogue: bias + layout-mapped store ----
#pragma unroll
    for (int ct = 0; ct < 2; ++ct) {
        const int col = colBase + wc * 32 + ct * 16 + m16;
        const float bv = bias[col];
#pragma unroll
        for (int j = 0; j < 8; ++j) {
            const int rg = rowBase + wr * 16 + hi * 8 + j;
            const float v = acc[ct].f[j] + bv;
            if (MODE == 0) {
                const int bb = rg >> 10;           // S = 1024
                const int ss = rg & 1023;
                const int hh = col / headDim;
                const int dd = col - hh * headDim;
                const size_t idx =
                    ((size_t)(bb * 4 + hh) * 1024 + ss) * outPitch + outOff + dd;
                outH[idx] = (_Float16)v;
            } else {
                outF[(size_t)rg * N + col] = v;
            }
        }
    }
}

// ------------------------------- attention ----------------------------------
// One block = 64 q-rows of one (b,h); 4 waves, each a 16-row q-tile.
// Flash-style online softmax over S=1024 keys, 32 keys/iteration.
// scores = (Q·K^T + QP·KP^T + QA·KA^T)/8 + mask ; ctx = softmax(scores)·V
__global__ __launch_bounds__(128) void attn_kernel(
    const _Float16* __restrict__ Qb,  const _Float16* __restrict__ Kb,
    const _Float16* __restrict__ Vb,  const _Float16* __restrict__ QPb,
    const _Float16* __restrict__ KPb, const _Float16* __restrict__ QAb,
    const _Float16* __restrict__ KAb, const float* __restrict__ mask,
    _Float16* __restrict__ ctx)
{
    __shared__ __align__(16) _Float16 VT[64][32];      // V^T tile [d][key]
    __shared__ __align__(16) _Float16 PT[4][32][16];   // per-wave P^T [key][qrow]

    const int tid  = threadIdx.x;
    const int lane = tid & 31;
    const int wave = tid >> 5;
    const int m16  = lane & 15;
    const int hi   = lane >> 4;
    const int klo  = hi * 8;

    const int qt = blockIdx.x & 15;
    const int bh = blockIdx.x >> 4;
    const int b  = bh >> 2;
    const int h  = bh & 3;
    const int qbase = qt * 64 + wave * 16;
    const size_t rowBH = (size_t)bh * 1024;

    const _Float16* Qr  = Qb  + (rowBH + qbase) * 64;
    const _Float16* QPr = QPb + (rowBH + qbase) * 64;
    const _Float16* QAr = QAb + (rowBH + qbase) * 32;
    const _Float16* Kr  = Kb  + rowBH * 64;
    const _Float16* KPr = KPb + rowBH * 64;
    const _Float16* KAr = KAb + rowBH * 32;
    const _Float16* Vr  = Vb  + rowBH * 64;
    const float*    Mr  = mask + (size_t)b * 1024 * 1024
                               + (size_t)(qbase + hi * 8) * 1024;

    // V staging coords: 2 segments/thread, each an 8-half contiguous load
    const int vkey0 = tid >> 3;            // 0..15
    const int vseg  = (tid & 7) * 8;       // feature segment 0..56

    // Query-side A fragments (loaded once): item(2 chunks)+pos(2)+attr(1)
    const v16h qf0 = load_afrag(Qr,  64, 0,  lane);
    const v16h qf1 = load_afrag(Qr,  64, 32, lane);
    const v16h pf0 = load_afrag(QPr, 64, 0,  lane);
    const v16h pf1 = load_afrag(QPr, 64, 32, lane);
    const v16h af0 = load_afrag(QAr, 32, 0,  lane);

    CFrag acc[4];
#pragma unroll
    for (int t = 0; t < 4; ++t)
#pragma unroll
        for (int j = 0; j < 8; ++j) acc[t].f[j] = 0.0f;

    float mrow[8], lrow[8];
#pragma unroll
    for (int j = 0; j < 8; ++j) { mrow[j] = -1e30f; lrow[j] = 0.0f; }

    const float LOG2E = 1.4426950408889634f;

    for (int kb = 0; kb < 1024; kb += 32) {
        // ---- cooperative stage of V^T tile (32 keys x 64 feats) ----
        // each thread: two 16B row-segment loads, scattered transpose stores
#pragma unroll
        for (int rseg = 0; rseg < 2; ++rseg) {
            const int key = vkey0 + rseg * 16;
            v8h vv = *(const v8h*)(Vr + (size_t)(kb + key) * 64 + vseg);
#pragma unroll
            for (int i = 0; i < 8; ++i) VT[vseg + i][key] = vv[i];
        }

        // ---- score sub-tiles: keys [kb..kb+15], [kb+16..kb+31] ----
        CFrag s0, s1;
        {
            v8f s = {};
            s = wmma_f16(qf0, load_bfrag_rows(Kr,  kb, 64, 0,  lane), s);
            s = wmma_f16(qf1, load_bfrag_rows(Kr,  kb, 64, 32, lane), s);
            s = wmma_f16(pf0, load_bfrag_rows(KPr, kb, 64, 0,  lane), s);
            s = wmma_f16(pf1, load_bfrag_rows(KPr, kb, 64, 32, lane), s);
            s = wmma_f16(af0, load_bfrag_rows(KAr, kb, 32, 0,  lane), s);
            s0.v = s;
        }
        {
            v8f s = {};
            s = wmma_f16(qf0, load_bfrag_rows(Kr,  kb + 16, 64, 0,  lane), s);
            s = wmma_f16(qf1, load_bfrag_rows(Kr,  kb + 16, 64, 32, lane), s);
            s = wmma_f16(pf0, load_bfrag_rows(KPr, kb + 16, 64, 0,  lane), s);
            s = wmma_f16(pf1, load_bfrag_rows(KPr, kb + 16, 64, 32, lane), s);
            s = wmma_f16(af0, load_bfrag_rows(KAr, kb + 16, 32, 0,  lane), s);
            s1.v = s;
        }

        // ---- scale by 1/sqrt(64) and add mask ----
        const float* M0 = Mr + kb + m16;
        const float* M1 = Mr + kb + 16 + m16;
#pragma unroll
        for (int j = 0; j < 8; ++j) {
            s0.f[j] = s0.f[j] * 0.125f + M0[(size_t)j * 1024];
            s1.f[j] = s1.f[j] * 0.125f + M1[(size_t)j * 1024];
        }

        // ---- online softmax update (reduce across 16-lane half groups) ----
        float scale[8];
#pragma unroll
        for (int j = 0; j < 8; ++j) {
            float mc = fmaxf(s0.f[j], s1.f[j]);
            mc = fmaxf(mc, __shfl_xor(mc, 1, 32));
            mc = fmaxf(mc, __shfl_xor(mc, 2, 32));
            mc = fmaxf(mc, __shfl_xor(mc, 4, 32));
            mc = fmaxf(mc, __shfl_xor(mc, 8, 32));
            const float mn = fmaxf(mrow[j], mc);
            scale[j] = exp2f((mrow[j] - mn) * LOG2E);
            mrow[j] = mn;
            s0.f[j] = exp2f((s0.f[j] - mn) * LOG2E);
            s1.f[j] = exp2f((s1.f[j] - mn) * LOG2E);
            float rs = s0.f[j] + s1.f[j];
            rs += __shfl_xor(rs, 1, 32);
            rs += __shfl_xor(rs, 2, 32);
            rs += __shfl_xor(rs, 4, 32);
            rs += __shfl_xor(rs, 8, 32);
            lrow[j] = lrow[j] * scale[j] + rs;
        }
#pragma unroll
        for (int t = 0; t < 4; ++t)
#pragma unroll
            for (int j = 0; j < 8; ++j) acc[t].f[j] *= scale[j];

        // ---- write P^T to LDS (one packed 16B store per sub-tile) ----
        v8h p0, p1;
#pragma unroll
        for (int j = 0; j < 8; ++j) { p0[j] = (_Float16)s0.f[j]; p1[j] = (_Float16)s1.f[j]; }
        *(v8h*)&PT[wave][m16][hi * 8]      = p0;
        *(v8h*)&PT[wave][16 + m16][hi * 8] = p1;

        __syncthreads();   // covers VT staging + own-wave PT stores

        // ---- P A-fragment (gather transpose from P^T) ----
        AFrag pf;
#pragma unroll
        for (int i = 0; i < 8; ++i) {
            pf.e[i]     = PT[wave][klo + i][m16];
            pf.e[8 + i] = PT[wave][klo + 16 + i][m16];
        }

        // ---- ctx += P @ V  (4 feature tiles of 16) ----
#pragma unroll
        for (int t = 0; t < 4; ++t) {
            AFrag vb;
            const _Float16* r = &VT[t * 16 + m16][hi * 8];
            vb.h[0] = *(const v8h*)(r);
            vb.h[1] = *(const v8h*)(r + 16);
            acc[t].v = wmma_f16(pf.v, vb.v, acc[t].v);
        }
        __syncthreads();   // before next iteration restages VT/PT
    }

    // ---- finalize: divide by l, merge heads, store f16 ctx [B,S,256] ----
#pragma unroll
    for (int j = 0; j < 8; ++j) {
        const float inv = 1.0f / lrow[j];
        const size_t row = (size_t)b * 1024 + qbase + hi * 8 + j;
#pragma unroll
        for (int t = 0; t < 4; ++t) {
            ctx[row * 256 + h * 64 + t * 16 + m16] = (_Float16)(acc[t].f[j] * inv);
        }
    }
}

// --------------------------- residual + LayerNorm ---------------------------
__global__ __launch_bounds__(32) void ln_kernel(
    const float* __restrict__ hidden, const float* __restrict__ resid,
    const float* __restrict__ gamma,  const float* __restrict__ beta,
    float* __restrict__ out)
{
    const int r  = blockIdx.x;
    const int c0 = threadIdx.x * 8;
    const float4* hp = (const float4*)(hidden + (size_t)r * 256 + c0);
    const float4* rp = (const float4*)(resid  + (size_t)r * 256 + c0);
    float4 h0 = hp[0], h1 = hp[1], r0 = rp[0], r1 = rp[1];
    float xv[8] = { h0.x + r0.x, h0.y + r0.y, h0.z + r0.z, h0.w + r0.w,
                    h1.x + r1.x, h1.y + r1.y, h1.z + r1.z, h1.w + r1.w };
    float s = 0.0f, s2 = 0.0f;
#pragma unroll
    for (int j = 0; j < 8; ++j) { s += xv[j]; s2 += xv[j] * xv[j]; }
#pragma unroll
    for (int off = 16; off >= 1; off >>= 1) {
        s  += __shfl_xor(s,  off, 32);
        s2 += __shfl_xor(s2, off, 32);
    }
    const float mu   = s * (1.0f / 256.0f);
    const float var  = s2 * (1.0f / 256.0f) - mu * mu;
    const float rstd = rsqrtf(var + 1e-12f);
#pragma unroll
    for (int j = 0; j < 8; ++j) {
        out[(size_t)r * 256 + c0 + j] =
            (xv[j] - mu) * rstd * gamma[c0 + j] + beta[c0 + j];
    }
}

// -------------------------------- launcher ----------------------------------
extern "C" void kernel_launch(void* const* d_in, const int* in_sizes, int n_in,
                              void* d_out, int out_size, void* d_ws, size_t ws_size,
                              hipStream_t stream)
{
    const float* x    = (const float*)d_in[0];
    const float* a0   = (const float*)d_in[1];
    const float* a1   = (const float*)d_in[2];
    const float* pe   = (const float*)d_in[3];
    const float* mask = (const float*)d_in[4];
    const float* Wq  = (const float*)d_in[5],  *bq  = (const float*)d_in[6];
    const float* Wk  = (const float*)d_in[7],  *bk  = (const float*)d_in[8];
    const float* Wv  = (const float*)d_in[9],  *bv  = (const float*)d_in[10];
    const float* Wqp = (const float*)d_in[11], *bqp = (const float*)d_in[12];
    const float* Wkp = (const float*)d_in[13], *bkp = (const float*)d_in[14];
    const float* Wq0 = (const float*)d_in[15], *bq0 = (const float*)d_in[16];
    const float* Wk0 = (const float*)d_in[17], *bk0 = (const float*)d_in[18];
    const float* Wq1 = (const float*)d_in[19], *bq1 = (const float*)d_in[20];
    const float* Wk1 = (const float*)d_in[21], *bk1 = (const float*)d_in[22];
    const float* Wd  = (const float*)d_in[23], *bd  = (const float*)d_in[24];
    const float* lng = (const float*)d_in[25], *lnb = (const float*)d_in[26];
    float* out = (float*)d_out;

    // ---- workspace carve-up (f16 elements; ~76 MB total) ----
    _Float16* H = (_Float16*)d_ws;
    size_t o = 0;
    auto nextH = [&](size_t n) { _Float16* p = H + o; o += n; return p; };
    _Float16 *WqB  = nextH(65536), *WkB  = nextH(65536), *WvB = nextH(65536);
    _Float16 *WqpB = nextH(65536), *WkpB = nextH(65536), *WdB = nextH(65536);
    _Float16 *Wq0B = nextH(4096),  *Wk0B = nextH(4096);
    _Float16 *Wq1B = nextH(4096),  *Wk1B = nextH(4096);
    const size_t BSH = 4194304;                     // B*S*HID = B*NH*S*HD
    _Float16 *Qb  = nextH(BSH), *Kb  = nextH(BSH), *Vb = nextH(BSH);
    _Float16 *QPb = nextH(BSH), *KPb = nextH(BSH);
    _Float16 *QAb = nextH(2097152), *KAb = nextH(2097152);  // B*NH*S*32
    _Float16 *CTXb = nextH(BSH);
    float* hid = (float*)(H + o);                   // 16 MB f32, 16B-aligned

    // ---- convert weights to f16 ----
    auto cv = [&](const float* s, _Float16* d, int n) {
        cvt_f32_f16<<<(n + 255) / 256, 256, 0, stream>>>(s, d, n);
    };
    cv(Wq, WqB, 65536);   cv(Wk, WkB, 65536);   cv(Wv, WvB, 65536);
    cv(Wqp, WqpB, 65536); cv(Wkp, WkpB, 65536); cv(Wd, WdB, 65536);
    cv(Wq0, Wq0B, 4096);  cv(Wk0, Wk0B, 4096);
    cv(Wq1, Wq1B, 4096);  cv(Wk1, Wk1B, 4096);

    const int M = 16384;                 // B*S
    dim3 gHID(4, 256), gA(1, 256), blk(256);
    // item / positional projections -> head-split [B,NH,S,64] f16
    gemm_proj<0,0><<<gHID, blk, 0, stream>>>(x,  nullptr, WqB,  bq,  Qb,  nullptr, M, 256, 256, 64, 64, 0);
    gemm_proj<0,0><<<gHID, blk, 0, stream>>>(x,  nullptr, WkB,  bk,  Kb,  nullptr, M, 256, 256, 64, 64, 0);
    gemm_proj<0,0><<<gHID, blk, 0, stream>>>(x,  nullptr, WvB,  bv,  Vb,  nullptr, M, 256, 256, 64, 64, 0);
    gemm_proj<0,0><<<gHID, blk, 0, stream>>>(pe, nullptr, WqpB, bqp, QPb, nullptr, M, 256, 256, 64, 64, 0);
    gemm_proj<0,0><<<gHID, blk, 0, stream>>>(pe, nullptr, WkpB, bkp, KPb, nullptr, M, 256, 256, 64, 64, 0);
    // attribute projections -> packed [B,NH,S,32] (qa0|qa1 concatenated)
    gemm_proj<0,0><<<gA, blk, 0, stream>>>(a0, nullptr, Wq0B, bq0, QAb, nullptr, M, 64, 64, 16, 32, 0);
    gemm_proj<0,0><<<gA, blk, 0, stream>>>(a0, nullptr, Wk0B, bk0, KAb, nullptr, M, 64, 64, 16, 32, 0);
    gemm_proj<0,0><<<gA, blk, 0, stream>>>(a1, nullptr, Wq1B, bq1, QAb, nullptr, M, 64, 64, 16, 32, 16);
    gemm_proj<0,0><<<gA, blk, 0, stream>>>(a1, nullptr, Wk1B, bk1, KAb, nullptr, M, 64, 64, 16, 32, 16);

    // fused 3-source attention with online softmax -> ctx [B,S,256] f16
    attn_kernel<<<1024, 128, 0, stream>>>(Qb, Kb, Vb, QPb, KPb, QAb, KAb, mask, CTXb);

    // output projection -> hidden f32 [M,256] (async global->LDS path)
    gemm_proj<1,1><<<gHID, blk, 0, stream>>>(nullptr, CTXb, WdB, bd, nullptr, hid, M, 256, 256, 64, 0, 0);

    // residual add + LayerNorm -> out f32
    ln_kernel<<<16384, 32, 0, stream>>>(hid, x, lng, lnb, out);
}
